// BiLSTM_CRF_10385230922256
// MI455X (gfx1250) — compile-verified, matrix operand order checked by reading, not run on hardware
//
#include <hip/hip_runtime.h>
#include <hip/hip_bf16.h>
#include <math.h>

// Problem constants (match reference)
#define B_   32
#define T_   512
#define E_   256
#define H_   256
#define G_   1024   // 4*H
#define C_   8
#define BOS_ 2
#define EOS_ 3

typedef __attribute__((ext_vector_type(16))) __bf16 v16bf;
typedef __attribute__((ext_vector_type(8)))  float  v8f;

// CDNA5 async global->LDS DMA path (ASYNCcnt-tracked), guarded so the file
// compiles on toolchains that lack the builtins.
#if defined(__HIP_DEVICE_COMPILE__) &&                                         \
    __has_builtin(__builtin_amdgcn_global_load_async_to_lds_b128) &&           \
    __has_builtin(__builtin_amdgcn_s_wait_asynccnt)
#define ASYNC_XG 1
// Builtin signature (from hipcc diagnostic): param0 = AS1 int4*, param1 = AS3 int4*
typedef int v4i_ __attribute__((vector_size(16)));
typedef __attribute__((address_space(1))) v4i_* gv4i_p;
typedef __attribute__((address_space(3))) v4i_* lv4i_p;
#else
#define ASYNC_XG 0
#endif

__device__ __forceinline__ float sigmoidf_(float x) { return 1.0f / (1.0f + __expf(-x)); }

__device__ __forceinline__ v8f wmma_bf16(v16bf a, v16bf b, v8f c) {
    // v_wmma_f32_16x16x32_bf16: D = A(16x32) * B(32x16) + C(16x16 f32)
    return __builtin_amdgcn_wmma_f32_16x16x32_bf16(false, a, false, b, (short)0, c, false, false);
}

// ---------------------------------------------------------------------------
// Pack W (N x K, row-major f32) into per-lane bf16 B-fragments for
// v_wmma_f32_16x16x32_bf16:  frag(ntile, ktile)[lane][i] = W[n, k]
//   n = ntile*16 + (lane & 15),  k = ktile*32 + (lane>>4)*16 + i
// Each lane's 16 bf16 (32 bytes) are contiguous -> clean b128 loads later.
// ---------------------------------------------------------------------------
__global__ __launch_bounds__(256) void pack_wT_bf16(const float* __restrict__ w,
                                                    __bf16* __restrict__ out,
                                                    int N, int K) {
    int tid = blockIdx.x * blockDim.x + threadIdx.x;
    int kt = K / 32;
    int total = (N / 16) * kt * 32;
    if (tid >= total) return;
    int lane  = tid & 31;
    int ktile = (tid >> 5) % kt;
    int ntile = (tid >> 5) / kt;
    int n  = ntile * 16 + (lane & 15);
    int k0 = ktile * 32 + (lane >> 4) * 16;
    const float* src = w + (size_t)n * K + k0;
    __bf16* dst = out + (size_t)tid * 16;
#pragma unroll
    for (int i = 0; i < 16; ++i) dst[i] = (__bf16)src[i];
}

// ---------------------------------------------------------------------------
// Fused embedding gather + input projection (both directions):
//   xg_{f,b}[b*T+t, :] = emb[tokens[b,t]] @ W_ih_{f,b}^T + b_ih_{f,b}
// GEMM: M = B*T = 16384, K = E = 256, N = 2048 (fwd|bwd).
// Block = 256 thr (8 waves): M-block 64 rows, all 128 N-tiles in-loop.
// ---------------------------------------------------------------------------
__global__ __launch_bounds__(256) void embed_gates(
        const int* __restrict__ tokens, const float* __restrict__ emb,
        const __bf16* __restrict__ wpk_f, const __bf16* __restrict__ wpk_b,
        const float* __restrict__ b_ih_f, const float* __restrict__ b_ih_b,
        float* __restrict__ xgf, float* __restrict__ xgb) {
    __shared__ __attribute__((aligned(32))) __bf16 Abf[64][E_ + 16]; // +16 bf16 pad
    int tid = threadIdx.x;
    int rowblk = blockIdx.x * 64;

    { // gather 64 embedding rows -> bf16 LDS tile (4 threads per row)
        int r  = tid >> 2;
        int cg = (tid & 3) * 64;
        int tok = tokens[rowblk + r];
        const float4* src = (const float4*)(emb + (size_t)tok * E_ + cg);
#pragma unroll
        for (int i = 0; i < 16; ++i) {
            float4 v = src[i];
            int c = cg + i * 4;
            Abf[r][c + 0] = (__bf16)v.x;
            Abf[r][c + 1] = (__bf16)v.y;
            Abf[r][c + 2] = (__bf16)v.z;
            Abf[r][c + 3] = (__bf16)v.w;
        }
    }
    __syncthreads();

    int wv    = tid >> 5;    // wave 0..7
    int lane  = tid & 31;
    int mh    = lane >> 4;   // K-half select for A frag / M-half for C
    int nlane = lane & 15;

    for (int mt = 0; mt < 4; ++mt) {
        v16bf a[8];
#pragma unroll
        for (int k = 0; k < 8; ++k)
            a[k] = *(const v16bf*)&Abf[mt * 16 + nlane][k * 32 + mh * 16];

        // dir-split loops: loop-invariant weight/bias/output pointers (no
        // per-iteration 64-bit cndmask selects)
#pragma unroll 1
        for (int half = 0; half < 2; ++half) {
            const __bf16* wpk  = half ? wpk_b : wpk_f;
            const float*  bih  = half ? b_ih_b : b_ih_f;
            float*        xg   = half ? xgb : xgf;
            for (int nt = 0; nt < 8; ++nt) {
                int nloc = wv + nt * 8;          // 0..63 within this direction
                v8f acc = {};
#pragma unroll
                for (int k = 0; k < 8; ++k) {
                    v16bf bf = *(const v16bf*)(wpk + (((size_t)nloc * 8 + k) * 32 + lane) * 16);
                    acc = wmma_bf16(a[k], bf, acc);
                }
                int n = nloc * 16 + nlane;
                float bias = bih[n];
#pragma unroll
                for (int r = 0; r < 8; ++r) {
                    int m = rowblk + mt * 16 + r + mh * 8;
                    xg[(size_t)m * G_ + n] = acc[r] + bias;
                }
            }
        }
    }
}

// ---------------------------------------------------------------------------
// Persistent LSTM scan: 1 workgroup per direction, 1024 thr = 32 waves.
// w_hh held bf16-resident in VGPRs (2 N-tiles x 8 K-frags per wave = 128 VGPRs)
// -> zero recurrent-weight traffic over the 512 sequential steps.
// xg[t] tiles (128 KB/step) double-buffered in LDS via async global->LDS DMA,
// issued 2 steps ahead; gate values are written back IN PLACE over the xg
// buffer (each (b,n) owned by one lane), so no extra LDS for gate exchange.
// LDS: 2*128K (xg/gates) + 32K (c) + 17K (h bf16) = 305 KB <= 320 KB/WGP.
// ---------------------------------------------------------------------------
__global__ __launch_bounds__(1024) void lstm_scan(
        const float* __restrict__ xgf, const float* __restrict__ xgb,
        const __bf16* __restrict__ wpk_hhf, const __bf16* __restrict__ wpk_hhb,
        const float* __restrict__ b_hh_f, const float* __restrict__ b_hh_b,
        const float* __restrict__ h0, const float* __restrict__ c0,
        float* __restrict__ hsf, float* __restrict__ hsb) {
    extern __shared__ char smem[];
    float*  xgbuf0 = (float*)smem;                                   // [32][1024]
    float*  xgbuf1 = (float*)(smem + (size_t)B_ * G_ * 4);           // [32][1024]
    float*  cst    = (float*)(smem + (size_t)2 * B_ * G_ * 4);       // [32][256]
    __bf16* hbf    = (__bf16*)(smem + (size_t)2 * B_ * G_ * 4 + (size_t)B_ * H_ * 4); // [32][272]
    const int HP = H_ + 16; // pad keeps 32B fragment alignment, rotates banks

    int dir = blockIdx.x;
    const float*  xg  = dir ? xgb : xgf;
    const __bf16* wpk = dir ? wpk_hhb : wpk_hhf;
    const float*  bhh = dir ? b_hh_b : b_hh_f;
    float*        hs  = dir ? hsb : hsf;

    int tid   = threadIdx.x;
    int wv    = tid >> 5;   // 0..31
    int lane  = tid & 31;
    int mh    = lane >> 4;
    int nlane = lane & 15;

    // Copy xg[time(step)] tile (32 rows x 4KB) into an LDS buffer.
    // Thread layout: row b = tid>>5, 8 consecutive 16B chunks per thread.
    auto issue_xg_copy = [&](int step, float* dstbuf) {
        if (step >= T_) return;
        int trow = dir ? (T_ - 1 - step) : step;
        int b     = tid >> 5;
        int inrow = (tid & 31) * 128;   // bytes within the 4KB row
        const char* gsrc = (const char*)(xg + ((size_t)b * T_ + trow) * G_) + inrow;
        char*       ldst = (char*)dstbuf + b * 4096 + inrow;
#if ASYNC_XG
#pragma unroll
        for (int j = 0; j < 8; ++j)
            __builtin_amdgcn_global_load_async_to_lds_b128(
                (gv4i_p)(gsrc + j * 16),
                (lv4i_p)(ldst + j * 16),
                0, 0);
#else
#pragma unroll
        for (int j = 0; j < 8; ++j)
            *(float4*)(ldst + j * 16) = *(const float4*)(gsrc + j * 16);
#endif
    };

    // Prologue: DMA the first two xg tiles while we set up state/weights.
    issue_xg_copy(0, xgbuf0);
    issue_xg_copy(1, xgbuf1);

    // init recurrent state
#pragma unroll
    for (int j = 0; j < 8; ++j) {
        int idx = tid * 8 + j;          // 0..8191 over (b, h)
        int b = idx >> 8, hh = idx & 255;
        hbf[b * HP + hh] = (__bf16)h0[(size_t)dir * B_ * H_ + idx];
        cst[b * H_ + hh] = c0[(size_t)dir * B_ * H_ + idx];
    }

    // VGPR-resident recurrent weights: wave wv owns N-tiles 2wv, 2wv+1
    v16bf Bf[2][8];
#pragma unroll
    for (int ntl = 0; ntl < 2; ++ntl) {
        int ntile = wv * 2 + ntl;
#pragma unroll
        for (int k = 0; k < 8; ++k)
            Bf[ntl][k] = *(const v16bf*)(wpk + (((size_t)ntile * 8 + k) * 32 + lane) * 16);
    }

    for (int t = 0; t < T_; ++t) {
        int trow = dir ? (T_ - 1 - t) : t;       // time index of this scan step
        float* gbuf = (t & 1) ? xgbuf1 : xgbuf0; // holds xg[trow]; becomes gates

        // Wait for this step's DMA: each copy = 8 async ops/wave, in-order
        // retirement, so <=8 outstanding means the older copy is complete.
#if ASYNC_XG
        if (t + 1 < T_) __builtin_amdgcn_s_wait_asynccnt(8);
        else            __builtin_amdgcn_s_wait_asynccnt(0);
#endif
        __syncthreads();   // DMA visible WGP-wide; prev step's hbf/cst done

        v8f acc[2][2];
#pragma unroll
        for (int mt = 0; mt < 2; ++mt)
#pragma unroll
            for (int ntl = 0; ntl < 2; ++ntl) { v8f z = {}; acc[mt][ntl] = z; }

#pragma unroll
        for (int mt = 0; mt < 2; ++mt) {
#pragma unroll
            for (int k = 0; k < 8; ++k) {
                v16bf a = *(const v16bf*)&hbf[(mt * 16 + nlane) * HP + k * 32 + mh * 16];
                acc[mt][0] = wmma_bf16(a, Bf[0][k], acc[mt][0]);
                acc[mt][1] = wmma_bf16(a, Bf[1][k], acc[mt][1]);
            }
        }

        // L2 prefetch a few steps ahead of the DMA engine
        if (t + 4 < T_) {
            int trow4 = dir ? (T_ - 1 - (t + 4)) : (t + 4);
            __builtin_prefetch(&xg[((size_t)(tid >> 5) * T_ + trow4) * G_ + (tid & 31) * 32], 0, 2);
        }

        // gates = h@W_hh^T + xg[t] + b_hh, written IN PLACE into gbuf
#pragma unroll
        for (int mt = 0; mt < 2; ++mt) {
#pragma unroll
            for (int ntl = 0; ntl < 2; ++ntl) {
                int n = (wv * 2 + ntl) * 16 + nlane;
                float bias = bhh[n];
#pragma unroll
                for (int r = 0; r < 8; ++r) {
                    int b = mt * 16 + r + mh * 8;
                    gbuf[b * G_ + n] = acc[mt][ntl][r] + gbuf[b * G_ + n] + bias;
                }
            }
        }
        __syncthreads();

        // elementwise LSTM cell update: each thread handles 8 (b,h) pairs
#pragma unroll
        for (int j = 0; j < 8; ++j) {
            int idx = tid * 8 + j;
            int b = idx >> 8, hh = idx & 255;
            float gi = gbuf[b * G_ + hh];
            float gf = gbuf[b * G_ + 256 + hh];
            float gg = gbuf[b * G_ + 512 + hh];
            float go = gbuf[b * G_ + 768 + hh];
            float c  = sigmoidf_(gf) * cst[b * H_ + hh] + sigmoidf_(gi) * tanhf(gg);
            float h  = sigmoidf_(go) * tanhf(c);
            cst[b * H_ + hh] = c;
            hbf[b * HP + hh] = (__bf16)h;
            hs[((size_t)trow * B_ + b) * H_ + hh] = h;
        }
        __syncthreads();   // gbuf fully consumed -> safe to refill

        // kick off DMA for step t+2 into the buffer just freed
        issue_xg_copy(t + 2, gbuf);
    }
}

// ---------------------------------------------------------------------------
// emissions[t,b,c] = [hs_f ; hs_b] . w_out[c] + b_out[c]   (C=8, tiny)
// ---------------------------------------------------------------------------
__global__ __launch_bounds__(256) void emissions_k(
        const float* __restrict__ hsf, const float* __restrict__ hsb,
        const float* __restrict__ w_out, const float* __restrict__ b_out,
        float* __restrict__ emis) {
    int tid = blockIdx.x * 256 + threadIdx.x;   // (t*B + b)*C + c
    if (tid >= T_ * B_ * C_) return;
    int c  = tid & 7;
    int tb = tid >> 3;
    const float* hf = hsf + (size_t)tb * H_;
    const float* hb = hsb + (size_t)tb * H_;
    const float* w  = w_out + (size_t)c * (2 * H_);
    float s = b_out[c];
#pragma unroll 4
    for (int j = 0; j < H_; ++j) s += hf[j] * w[j];
#pragma unroll 4
    for (int j = 0; j < H_; ++j) s += hb[j] * w[H_ + j];
    emis[tid] = s;
}

// ---------------------------------------------------------------------------
// Viterbi (C=8) + backtrace in one block: thread = b*8 + j.
// ---------------------------------------------------------------------------
__global__ __launch_bounds__(256) void viterbi_k(
        const float* __restrict__ emis, const unsigned char* __restrict__ mask,
        const float* __restrict__ trans, int* __restrict__ bps,
        float* __restrict__ out) {
    __shared__ float alph[B_][C_];
    __shared__ float tr[C_][C_];
    __shared__ int   lastS[B_];
    __shared__ float scoreS[B_];
    int tid = threadIdx.x;
    if (tid < C_ * C_) tr[tid >> 3][tid & 7] = trans[tid];
    int b = tid >> 3, j = tid & 7;
    __syncthreads();
    alph[b][j] = tr[BOS_][j] + emis[b * C_ + j];
    __syncthreads();

    for (int t = 1; t < T_; ++t) {
        float e = emis[((size_t)t * B_ + b) * C_ + j];
        float best = alph[b][0] + tr[0][j];
        int bi = 0;
#pragma unroll
        for (int i = 1; i < C_; ++i) {
            float v = alph[b][i] + tr[i][j];
            if (v > best) { best = v; bi = i; }   // first-max tie break (argmax)
        }
        unsigned char m = mask[b * T_ + t];
        float na = m ? (best + e) : alph[b][j];
        int   bp = m ? bi : j;
        bps[((size_t)(t - 1) * B_ + b) * C_ + j] = bp;
        __syncthreads();
        alph[b][j] = na;
        __syncthreads();
    }

    alph[b][j] += tr[j][EOS_];
    __syncthreads();
    if (j == 0) {
        float best = alph[b][0];
        int bi = 0;
#pragma unroll
        for (int i = 1; i < C_; ++i)
            if (alph[b][i] > best) { best = alph[b][i]; bi = i; }
        scoreS[b] = best;
        lastS[b] = bi;
    }
    __syncthreads();
    if (tid < B_) {
        int bb = tid;
        out[bb] = scoreS[bb];
        int cur = lastS[bb];
        float* path = out + B_ + (size_t)bb * T_;
        path[T_ - 1] = mask[bb * T_ + (T_ - 1)] ? (float)cur : 0.0f;
        for (int t = T_ - 2; t >= 0; --t) {
            cur = bps[((size_t)t * B_ + bb) * C_ + cur];
            path[t] = mask[bb * T_ + t] ? (float)cur : 0.0f;
        }
    }
}

// ---------------------------------------------------------------------------
extern "C" void kernel_launch(void* const* d_in, const int* in_sizes, int n_in,
                              void* d_out, int out_size, void* d_ws, size_t ws_size,
                              hipStream_t stream) {
    (void)in_sizes; (void)n_in; (void)out_size; (void)ws_size;
    const int*           tokens = (const int*)d_in[0];
    const unsigned char* mask   = (const unsigned char*)d_in[1];
    const float* emb    = (const float*)d_in[2];
    const float* w_ih_f = (const float*)d_in[3];
    const float* w_hh_f = (const float*)d_in[4];
    const float* b_ih_f = (const float*)d_in[5];
    const float* b_hh_f = (const float*)d_in[6];
    const float* w_ih_b = (const float*)d_in[7];
    const float* w_hh_b = (const float*)d_in[8];
    const float* b_ih_b = (const float*)d_in[9];
    const float* b_hh_b = (const float*)d_in[10];
    const float* w_out  = (const float*)d_in[11];
    const float* b_out  = (const float*)d_in[12];
    const float* trans  = (const float*)d_in[13];
    const float* h0     = (const float*)d_in[14];
    const float* c0     = (const float*)d_in[15];
    float* out = (float*)d_out;

    // workspace carve-up (~163 MB total)
    char* ws = (char*)d_ws;
    size_t off = 0;
    auto alloc = [&](size_t bytes) -> char* {
        char* p = ws + off;
        off += (bytes + 255) & ~(size_t)255;
        return p;
    };
    __bf16* wpk_ihf = (__bf16*)alloc((size_t)G_ * E_ * 2);
    __bf16* wpk_ihb = (__bf16*)alloc((size_t)G_ * E_ * 2);
    __bf16* wpk_hhf = (__bf16*)alloc((size_t)G_ * H_ * 2);
    __bf16* wpk_hhb = (__bf16*)alloc((size_t)G_ * H_ * 2);
    float*  xgf  = (float*)alloc((size_t)B_ * T_ * G_ * 4);
    float*  xgb  = (float*)alloc((size_t)B_ * T_ * G_ * 4);
    float*  hsf  = (float*)alloc((size_t)T_ * B_ * H_ * 4);
    float*  hsb  = (float*)alloc((size_t)T_ * B_ * H_ * 4);
    float*  emis = (float*)alloc((size_t)T_ * B_ * C_ * 4);
    int*    bps  = (int*)alloc((size_t)(T_ - 1) * B_ * C_ * 4);

    // 1) pack weights into WMMA B-fragment layout (bf16)
    int packThreads = (G_ / 16) * (E_ / 32) * 32;   // 16384
    dim3 pg((packThreads + 255) / 256), pb(256);
    pack_wT_bf16<<<pg, pb, 0, stream>>>(w_ih_f, wpk_ihf, G_, E_);
    pack_wT_bf16<<<pg, pb, 0, stream>>>(w_ih_b, wpk_ihb, G_, E_);
    pack_wT_bf16<<<pg, pb, 0, stream>>>(w_hh_f, wpk_hhf, G_, H_);
    pack_wT_bf16<<<pg, pb, 0, stream>>>(w_hh_b, wpk_hhb, G_, H_);

    // 2) embedding gather + input projection (both directions), WMMA bf16
    embed_gates<<<dim3((B_ * T_) / 64), dim3(256), 0, stream>>>(
        tokens, emb, wpk_ihf, wpk_ihb, b_ih_f, b_ih_b, xgf, xgb);

    // 3) persistent bidirectional LSTM scan (1 WGP per direction)
    size_t scan_smem = (size_t)2 * B_ * G_ * 4      // xg double buffer / gates
                     + (size_t)B_ * H_ * 4          // c state
                     + (size_t)B_ * (H_ + 16) * 2;  // h state (bf16, padded)
    lstm_scan<<<dim3(2), dim3(1024), scan_smem, stream>>>(
        xgf, xgb, wpk_hhf, wpk_hhb, b_hh_f, b_hh_b, h0, c0, hsf, hsb);

    // 4) emissions
    emissions_k<<<dim3((T_ * B_ * C_) / 256), dim3(256), 0, stream>>>(
        hsf, hsb, w_out, b_out, emis);

    // 5) Viterbi + backtrace
    viterbi_k<<<dim3(1), dim3(256), 0, stream>>>(emis, mask, trans, bps, out);
}
